// BridgeTowerSelfAttention_42786464203086
// MI455X (gfx1250) — matrix operand order repaired and checked by reading, not compiled, not run on hardware
//
#include <hip/hip_runtime.h>
#include <hip/hip_bf16.h>

// ---------------------------------------------------------------------------
// Types for CDNA5 WMMA (wave32, 16x16x32 bf16 -> f32)
// ---------------------------------------------------------------------------
typedef __attribute__((ext_vector_type(16))) __bf16 v16bf;
typedef __attribute__((ext_vector_type(8)))  __bf16 v8bf;
typedef __attribute__((ext_vector_type(4)))  __bf16 v4bf;
typedef __attribute__((ext_vector_type(8)))  float  v8f;

#define WMMA_BF16(a, b, c) \
  __builtin_amdgcn_wmma_f32_16x16x32_bf16(false, (a), false, (b), (short)0, (c), false, false)

// ---------------------------------------------------------------------------
// Fragment loaders (sources row-major: A indexed [m][k], B indexed [n][k]).
// A-matrix 16x32 bf16 (ISA 7.12.2): lane holds row m=lane&15;
//   e=0..7  -> k = (lane>>4)*8 + e ; e=8..15 -> k = 16 + (lane>>4)*8 + (e-8)
// B-matrix 32x16 bf16: lane holds col n=lane&15; same k order as A:
//   e=0..7  -> k = (lane>>4)*8 + e ; e=8..15 -> k = 16 + (lane>>4)*8 + (e-8)
// ---------------------------------------------------------------------------
__device__ __forceinline__ v16bf frag_a_bf16(const __bf16* base, int ld) {
  const int lane = threadIdx.x & 31;
  const __bf16* p = base + (lane & 15) * ld + ((lane >> 4) << 3);
  v8bf lo = *(const v8bf*)p;
  v8bf hi = *(const v8bf*)(p + 16);
  v16bf r;
#pragma unroll
  for (int i = 0; i < 8; ++i) { r[i] = lo[i]; r[i + 8] = hi[i]; }
  return r;
}

__device__ __forceinline__ v16bf frag_b_bf16(const __bf16* base, int ld) {
  const int lane = threadIdx.x & 31;
  const __bf16* p = base + (lane & 15) * ld + ((lane >> 4) << 3);
  v8bf lo = *(const v8bf*)p;
  v8bf hi = *(const v8bf*)(p + 16);
  v16bf r;
#pragma unroll
  for (int i = 0; i < 8; ++i) { r[i] = lo[i]; r[i + 8] = hi[i]; }
  return r;
}

// ---------------------------------------------------------------------------
// Kernel 0a: bulk f32 -> bf16 conversion (vectorized 4 elems/thread).
// ---------------------------------------------------------------------------
__global__ void cvt_bf16_kernel(const float* __restrict__ src,
                                __bf16* __restrict__ dst, int n4) {
  const int i = blockIdx.x * blockDim.x + threadIdx.x;
  if (i < n4) {
    const float4 v = ((const float4*)src)[i];
    v4bf o;
    o[0] = (__bf16)v.x; o[1] = (__bf16)v.y; o[2] = (__bf16)v.z; o[3] = (__bf16)v.w;
    ((v4bf*)dst)[i] = o;
  }
}

// ---------------------------------------------------------------------------
// Kernel 0b: dist_emb (2047x64 f32) -> bf16 padded to 2048 rows (row 2047
// zeroed; it can be touched by the Toeplitz bias matmul overshoot).
// ---------------------------------------------------------------------------
__global__ void cvt_emb_kernel(const float* __restrict__ emb, __bf16* __restrict__ out) {
  const int idx = blockIdx.x * blockDim.x + threadIdx.x;  // 0 .. 2048*64-1
  const int row = idx >> 6;
  const float v = (row < 2047) ? emb[idx] : 0.0f;
  out[idx] = (__bf16)v;
}

// ---------------------------------------------------------------------------
// Kernel 1: QKV projection, pure bf16. Y = X @ W^T + b.
// X:[4096,1024] bf16, W:[3][1024,1024] bf16 (row-major, row = out-feature).
// 128x128 output tile per block; 4 waves, each owning a 64x64 quadrant
// (16 WMMAs per k-step against 8 fragment loads -> ~16 MAC/byte).
// z = 0/1/2 selects Q/K/V. Q,K stored bf16 [b,h,s,d]; V stored bf16 [b,h,d,s]
// (pre-transposed so the attention kernel reads V directly as B-fragments).
// ---------------------------------------------------------------------------
__global__ __launch_bounds__(128)
void qkv_proj_kernel(const __bf16* __restrict__ X, const __bf16* __restrict__ Wall,
                     const float* __restrict__ B0, const float* __restrict__ B1,
                     const float* __restrict__ B2,
                     __bf16* __restrict__ qbf, __bf16* __restrict__ kbf,
                     __bf16* __restrict__ vbf) {
  const int m0 = blockIdx.x * 128;
  const int n0 = blockIdx.y * 128;
  const int z  = blockIdx.z;
  const __bf16* W  = Wall + (size_t)z * 1024 * 1024;
  const float*  Bz = (z == 0) ? B0 : (z == 1) ? B1 : B2;

  const int wave = threadIdx.x >> 5;
  const int lane = threadIdx.x & 31;
  const int mo = (wave & 1) * 64;
  const int no = (wave >> 1) * 64;

  v8f acc[4][4] = {};
  for (int k = 0; k < 1024; k += 32) {
    v16bf a[4], b[4];
#pragma unroll
    for (int t = 0; t < 4; ++t)
      a[t] = frag_a_bf16(X + (size_t)(m0 + mo + t * 16) * 1024 + k, 1024);
#pragma unroll
    for (int t = 0; t < 4; ++t)
      b[t] = frag_b_bf16(W + (size_t)(n0 + no + t * 16) * 1024 + k, 1024);
#pragma unroll
    for (int mt = 0; mt < 4; ++mt)
#pragma unroll
      for (int nt = 0; nt < 4; ++nt)
        acc[mt][nt] = WMMA_BF16(a[mt], b[nt], acc[mt][nt]);
  }

  const int lg = lane >> 4, ln = lane & 15;
#pragma unroll
  for (int mt = 0; mt < 4; ++mt) {
#pragma unroll
    for (int nt = 0; nt < 4; ++nt) {
      const int n = n0 + no + nt * 16 + ln;            // feature index 0..1023
      const float bias = Bz[n];
      const int h = n >> 6, d = n & 63;
#pragma unroll
      for (int i = 0; i < 8; ++i) {
        const int m = m0 + mo + mt * 16 + i + lg * 8;  // row index 0..4095
        const int bb = m >> 10, s = m & 1023;
        const float y = acc[mt][nt][i] + bias;
        if (z == 0)
          qbf[(((size_t)(bb * 16 + h) * 1024) + s) * 64 + d] = (__bf16)y;
        else if (z == 1)
          kbf[(((size_t)(bb * 16 + h) * 1024) + s) * 64 + d] = (__bf16)y;
        else
          vbf[(((size_t)(bb * 16 + h) * 64) + d) * 1024 + s] = (__bf16)y;
      }
    }
  }
}

// ---------------------------------------------------------------------------
// Kernel 2: flash attention with relative_key_query bias.
//   bias[l,r] = (q[l] + k[r]) . emb[l-r+1023]
//   Tq[l,m] = Q @ emb[jb+m]^T , Tk[r,m] = K @ emb[jb+m]^T  (WMMA GEMMs),
//   score = (QK^T + Tq[l, ll-rl+63] + Tk[r, ll-rl+63]) / 8, online softmax.
// Block = 128 threads (4 waves); wave w owns query rows [w*16, w*16+16).
// ---------------------------------------------------------------------------
__global__ __launch_bounds__(128)
void attn_kernel(const __bf16* __restrict__ qbf, const __bf16* __restrict__ kbf,
                 const __bf16* __restrict__ vbf, const __bf16* __restrict__ embbf,
                 float* __restrict__ out) {
  __shared__ __bf16 Tq[64][136];   // bias tables (m in [0,128), padded stride)
  __shared__ __bf16 Tk[64][136];
  __shared__ __bf16 Pt[64][72];    // probs tile, A-fragment source

  const int bh = blockIdx.y;            // b*16 + h
  const int l0 = blockIdx.x * 64;
  const int wave = threadIdx.x >> 5;
  const int lane = threadIdx.x & 31;
  const int lg = lane >> 4, ln = lane & 15;
  const int mrow = wave * 16;

  const __bf16* q  = qbf + (size_t)bh * 1024 * 64;   // [s][d]
  const __bf16* k  = kbf + (size_t)bh * 1024 * 64;   // [s][d]
  const __bf16* vt = vbf + (size_t)bh * 64 * 1024;   // [d][s] (pre-transposed)

  v8f O[4] = {};
  float run_max[8], run_sum[8];
#pragma unroll
  for (int i = 0; i < 8; ++i) { run_max[i] = -3.0e38f; run_sum[i] = 0.0f; }

  // Q fragments for this wave's 16 rows: reused across the whole key sweep.
  const v16bf qa0 = frag_a_bf16(q + (size_t)(l0 + mrow) * 64, 64);
  const v16bf qa1 = frag_a_bf16(q + (size_t)(l0 + mrow) * 64 + 32, 64);

  for (int r0 = 0; r0 < 1024; r0 += 64) {
    const int jb = l0 - r0 + 960;       // emb row base; in [0, 1920]

    // --- bias tables Tq (64x128) and Tk (64x128) via WMMA --------------------
    const v16bf ka0 = frag_a_bf16(k + (size_t)(r0 + mrow) * 64, 64);
    const v16bf ka1 = frag_a_bf16(k + (size_t)(r0 + mrow) * 64 + 32, 64);
#pragma unroll
    for (int nt = 0; nt < 8; ++nt) {
      const v16bf eb0 = frag_b_bf16(embbf + (size_t)(jb + nt * 16) * 64, 64);
      const v16bf eb1 = frag_b_bf16(embbf + (size_t)(jb + nt * 16) * 64 + 32, 64);
      v8f tq = {};
      tq = WMMA_BF16(qa0, eb0, tq);
      tq = WMMA_BF16(qa1, eb1, tq);
      v8f tk = {};
      tk = WMMA_BF16(ka0, eb0, tk);
      tk = WMMA_BF16(ka1, eb1, tk);
      const int col = nt * 16 + ln;
#pragma unroll
      for (int i = 0; i < 8; ++i) {
        Tq[mrow + i + lg * 8][col] = (__bf16)tq[i];
        Tk[mrow + i + lg * 8][col] = (__bf16)tk[i];
      }
    }
    __syncthreads();                    // Tk is read cross-wave below

    // --- scores: QK^T + Toeplitz-gathered biases ----------------------------
    v8f sc[4];
#pragma unroll
    for (int nt = 0; nt < 4; ++nt) {
      const v16bf kb0 = frag_b_bf16(k + (size_t)(r0 + nt * 16) * 64, 64);
      const v16bf kb1 = frag_b_bf16(k + (size_t)(r0 + nt * 16) * 64 + 32, 64);
      v8f s = {};
      s = WMMA_BF16(qa0, kb0, s);
      s = WMMA_BF16(qa1, kb1, s);
      const int rl = nt * 16 + ln;      // key index within the 64-tile
#pragma unroll
      for (int i = 0; i < 8; ++i) {
        const int ll = mrow + i + lg * 8;
        const int m = ll - rl + 63;     // 0..126
        s[i] = (s[i] + (float)Tq[ll][m] + (float)Tk[rl][m]) * 0.125f;
      }
      sc[nt] = s;
    }

    // --- in-wave online softmax (rows live in 16-lane halves) ---------------
    float scale_i[8];
#pragma unroll
    for (int i = 0; i < 8; ++i) {
      float mx = fmaxf(fmaxf(sc[0][i], sc[1][i]), fmaxf(sc[2][i], sc[3][i]));
      mx = fmaxf(mx, __shfl_xor(mx, 1, 32));
      mx = fmaxf(mx, __shfl_xor(mx, 2, 32));
      mx = fmaxf(mx, __shfl_xor(mx, 4, 32));
      mx = fmaxf(mx, __shfl_xor(mx, 8, 32));
      const float nm = fmaxf(run_max[i], mx);
      const float ss = __expf(run_max[i] - nm);
      run_max[i] = nm;
      float ps = 0.0f;
#pragma unroll
      for (int nt = 0; nt < 4; ++nt) {
        const float p = __expf(sc[nt][i] - nm);
        sc[nt][i] = p;
        ps += p;
      }
      ps += __shfl_xor(ps, 1, 32);
      ps += __shfl_xor(ps, 2, 32);
      ps += __shfl_xor(ps, 4, 32);
      ps += __shfl_xor(ps, 8, 32);
      run_sum[i] = run_sum[i] * ss + ps;
      scale_i[i] = ss;
    }

    // --- P to LDS (bf16), then O = O*scale + P @ V --------------------------
#pragma unroll
    for (int nt = 0; nt < 4; ++nt) {
      const int col = nt * 16 + ln;
#pragma unroll
      for (int i = 0; i < 8; ++i)
        Pt[mrow + i + lg * 8][col] = (__bf16)sc[nt][i];
    }
    const v16bf pa0 = frag_a_bf16(&Pt[mrow][0], 72);
    const v16bf pa1 = frag_a_bf16(&Pt[mrow][0] + 32, 72);
#pragma unroll
    for (int dt = 0; dt < 4; ++dt) {
      const v16bf vb0 = frag_b_bf16(vt + (size_t)(dt * 16) * 1024 + r0, 1024);
      const v16bf vb1 = frag_b_bf16(vt + (size_t)(dt * 16) * 1024 + r0 + 32, 1024);
#pragma unroll
      for (int i = 0; i < 8; ++i) O[dt][i] *= scale_i[i];
      O[dt] = WMMA_BF16(pa0, vb0, O[dt]);
      O[dt] = WMMA_BF16(pa1, vb1, O[dt]);
    }
    __syncthreads();                    // protect Tq/Tk for next iteration
  }

  // --- epilogue: normalize and store ctx [B, S, H*D] f32 --------------------
  const int b = bh >> 4, h = bh & 15;
#pragma unroll
  for (int dt = 0; dt < 4; ++dt) {
    const int d = dt * 16 + ln;
#pragma unroll
    for (int i = 0; i < 8; ++i) {
      const int ll = mrow + i + lg * 8;
      const float inv = 1.0f / run_sum[i];
      out[((size_t)b * 1024 + (l0 + ll)) * 1024 + h * 64 + d] = O[dt][i] * inv;
    }
  }
}

// ---------------------------------------------------------------------------
// Host launcher
// ---------------------------------------------------------------------------
extern "C" void kernel_launch(void* const* d_in, const int* in_sizes, int n_in,
                              void* d_out, int out_size, void* d_ws, size_t ws_size,
                              hipStream_t stream) {
  (void)in_sizes; (void)n_in; (void)out_size; (void)ws_size;
  const float* hs = (const float*)d_in[0];
  const float* wq = (const float*)d_in[1];
  const float* bq = (const float*)d_in[2];
  const float* wk = (const float*)d_in[3];
  const float* bk = (const float*)d_in[4];
  const float* wv = (const float*)d_in[5];
  const float* bv = (const float*)d_in[6];
  const float* de = (const float*)d_in[7];
  float* out = (float*)d_out;

  char* ws = (char*)d_ws;
  const size_t MB = 1024 * 1024;
  __bf16* qbf = (__bf16*)(ws);                 // 8 MB   [b,h,s,d]
  __bf16* kbf = (__bf16*)(ws + 8  * MB);       // 8 MB   [b,h,s,d]
  __bf16* vbf = (__bf16*)(ws + 16 * MB);       // 8 MB   [b,h,d,s]
  __bf16* ebf = (__bf16*)(ws + 24 * MB);       // 256 KB (2048x64)
  __bf16* xbf = (__bf16*)(ws + 25 * MB);       // 8 MB   hidden_states bf16
  __bf16* wbf = (__bf16*)(ws + 33 * MB);       // 6 MB   wq|wk|wv bf16

  // f32 -> bf16 staging passes (streaming, HBM-bound, ~µs)
  cvt_bf16_kernel<<<4096, 256, 0, stream>>>(hs, xbf, 4 * 1024 * 256);
  cvt_bf16_kernel<<<1024, 256, 0, stream>>>(wq, wbf + 0 * MB, 1024 * 256);
  cvt_bf16_kernel<<<1024, 256, 0, stream>>>(wk, wbf + 1 * MB, 1024 * 256);
  cvt_bf16_kernel<<<1024, 256, 0, stream>>>(wv, wbf + 2 * MB, 1024 * 256);
  cvt_emb_kernel<<<512, 256, 0, stream>>>(de, ebf);

  qkv_proj_kernel<<<dim3(32, 8, 3), 128, 0, stream>>>(xbf, wbf, bq, bk, bv,
                                                      qbf, kbf, vbf);
  attn_kernel<<<dim3(16, 64), 128, 0, stream>>>(qbf, kbf, vbf, ebf, out);
}
